// CPC_31696858644593
// MI455X (gfx1250) — matrix operand order repaired and checked by reading, not compiled, not run on hardware
//
#include <hip/hip_runtime.h>
#include <hip/hip_bf16.h>
#include <math.h>

// ---------------------------------------------------------------------------
// Problem: N=8192, IN=512, Z=128, C=64, H=50 (padded to 64).
// Strategy:
//   fx = relu(x@W1+b1)@W2+b2        -> bf16 WMMA GEMMs (K=512, K=64)
//   fz = z@Wz+bz                    -> bf16 WMMA GEMM  (K=128)
//   bucket rows by category (histogram + prefix scan + scatter)
//   per category k:  U = Fx_k @ Ws[k] (WMMA), T_i = softplus(U_i . fz_i),
//                    negT_i = mean_j softplus(U_i . Fz_k_j)  (WMMA + mask)
//   out[i] = log(T+eps) - log(negT+eps)
// ---------------------------------------------------------------------------

typedef __attribute__((ext_vector_type(16))) __bf16 v16bf;
typedef __attribute__((ext_vector_type(8)))  __bf16 v8bf;
typedef __attribute__((ext_vector_type(8)))  float  v8f;

#define N_TOT   8192
#define IN_DIM  512
#define Z_DIM   128
#define C_DIM   64
#define H_DIM   50
#define H_PAD   64
#define EPSV    1e-8f

__device__ __forceinline__ float softplus_f(float v) {
  if (v >  20.f) return v;
  if (v < -20.f) return __expf(v);
  return log1pf(__expf(v));
}

// A fragment (16x32 bf16) from row-major LDS tile. Per ISA 7.12.2:
// lane = hi*16 + m ; elements e<8 -> K = k0+hi*8+e ; e>=8 -> K = k0+16+hi*8+(e-8)
__device__ __forceinline__ v16bf frag_a(const __bf16* A, int lda, int m0, int k0, int lane) {
  const int m = lane & 15, hi = lane >> 4;
  const __bf16* p = A + (m0 + m) * lda + k0 + hi * 8;
  v8bf x0 = *(const v8bf*)(p);
  v8bf x1 = *(const v8bf*)(p + 16);
  v16bf r;
#pragma unroll
  for (int e = 0; e < 8; ++e) { r[e] = x0[e]; r[e + 8] = x1[e]; }
  return r;
}

// B fragment (32x16 bf16) from LDS holding B^T row-major (Bt[n][k]).
// Per ISA: lane = hi*16 + n ; elements e<8 -> K = k0+hi*8+e ; e>=8 -> K = k0+16+hi*8+(e-8)
__device__ __forceinline__ v16bf frag_bt(const __bf16* Bt, int ldb, int n0, int k0, int lane) {
  const int n = lane & 15, hi = lane >> 4;
  const __bf16* p = Bt + (n0 + n) * ldb + k0 + hi * 8;
  v8bf x0 = *(const v8bf*)(p);
  v8bf x1 = *(const v8bf*)(p + 16);
  v16bf r;
#pragma unroll
  for (int e = 0; e < 8; ++e) { r[e] = x0[e]; r[e + 8] = x1[e]; }
  return r;
}

__device__ __forceinline__ v8f wmma_bf16(v8f c, v16bf a, v16bf b) {
  return __builtin_amdgcn_wmma_f32_16x16x32_bf16(false, a, false, b, (short)0, c, false, false);
}

// ---------------------------------------------------------------------------
// fx = relu(x @ W1 + b1) @ W2 + b2, H padded 50->64 with zeros.
// 256 threads (8 waves), 64 rows per block.
// ---------------------------------------------------------------------------
__global__ __launch_bounds__(256) void k_fx(const float* __restrict__ x,
                                            const float* __restrict__ W1,
                                            const float* __restrict__ b1,
                                            const float* __restrict__ W2,
                                            const float* __restrict__ b2,
                                            float* __restrict__ fx) {
  __shared__ __bf16 sW1t[H_PAD * IN_DIM];  // W1^T padded: [h][in]   64 KB
  __shared__ __bf16 sX[64 * 128];          // x K-chunk               16 KB
  __shared__ __bf16 sHid[64 * H_PAD];      // relu hidden              8 KB
  __shared__ __bf16 sW2t[Z_DIM * H_PAD];   // W2^T padded: [z][h]     16 KB
  __shared__ float  sB1[H_PAD];
  __shared__ float  sB2[Z_DIM];

  const int tid = threadIdx.x;
  const int lane = tid & 31;
  const int wave = tid >> 5;
  const int hi = lane >> 4, nl = lane & 15;
  const int r0 = blockIdx.x * 64;

  for (int i = tid; i < H_PAD * IN_DIM; i += 256) {
    int h = i / IN_DIM, kin = i % IN_DIM;
    sW1t[i] = (h < H_DIM) ? (__bf16)W1[kin * H_DIM + h] : (__bf16)0.f;
  }
  for (int i = tid; i < Z_DIM * H_PAD; i += 256) {
    int zc = i / H_PAD, h = i % H_PAD;
    sW2t[i] = (h < H_DIM) ? (__bf16)W2[h * Z_DIM + zc] : (__bf16)0.f;
  }
  if (tid < H_PAD) sB1[tid] = (tid < H_DIM) ? b1[tid] : 0.f;
  if (tid < Z_DIM) sB2[tid] = b2[tid];

  // GEMM1: hidden[64][64] = x[64][512] @ W1p[512][64]; 16 tiles, 2/wave
  v8f acc0 = {}, acc1 = {};
  const int t0 = wave * 2, t1 = t0 + 1;
  const int m0a = (t0 >> 2) * 16, n0a = (t0 & 3) * 16;
  const int m1a = (t1 >> 2) * 16, n1a = (t1 & 3) * 16;
  for (int kc = 0; kc < IN_DIM; kc += 128) {
    __syncthreads();
    for (int i = tid; i < 64 * 128; i += 256) {
      int r = i >> 7, kk = i & 127;
      sX[i] = (__bf16)x[(size_t)(r0 + r) * IN_DIM + kc + kk];
    }
    __syncthreads();
#pragma unroll
    for (int kk = 0; kk < 128; kk += 32) {
      acc0 = wmma_bf16(acc0, frag_a(sX, 128, m0a, kk, lane),
                             frag_bt(sW1t, IN_DIM, n0a, kc + kk, lane));
      acc1 = wmma_bf16(acc1, frag_a(sX, 128, m1a, kk, lane),
                             frag_bt(sW1t, IN_DIM, n1a, kc + kk, lane));
    }
  }
  __syncthreads();
#pragma unroll
  for (int tt = 0; tt < 2; ++tt) {
    v8f a = tt ? acc1 : acc0;
    int m0 = tt ? m1a : m0a, n0 = tt ? n1a : n0a;
#pragma unroll
    for (int r = 0; r < 8; ++r) {
      int row = m0 + r + hi * 8, col = n0 + nl;
      float v = a[r] + sB1[col];
      sHid[row * H_PAD + col] = (__bf16)(v > 0.f ? v : 0.f);
    }
  }
  __syncthreads();

  // GEMM2: fx[64][128] = hidden[64][64] @ W2p[64][128]; 32 tiles, 4/wave
#pragma unroll
  for (int tt = 0; tt < 4; ++tt) {
    int t = wave * 4 + tt;
    int m0 = (t >> 3) * 16, n0 = (t & 7) * 16;
    v8f c = {};
#pragma unroll
    for (int kk = 0; kk < H_PAD; kk += 32)
      c = wmma_bf16(c, frag_a(sHid, H_PAD, m0, kk, lane),
                       frag_bt(sW2t, H_PAD, n0, kk, lane));
#pragma unroll
    for (int r = 0; r < 8; ++r) {
      int row = r0 + m0 + r + hi * 8, col = n0 + nl;
      fx[(size_t)row * Z_DIM + col] = c[r] + sB2[col];
    }
  }
}

// ---------------------------------------------------------------------------
// fz = z @ Wz + bz
// ---------------------------------------------------------------------------
__global__ __launch_bounds__(256) void k_fz(const float* __restrict__ z,
                                            const float* __restrict__ Wz,
                                            const float* __restrict__ bz,
                                            float* __restrict__ fz) {
  __shared__ __bf16 sWzt[Z_DIM * Z_DIM];  // Wz^T: [zo][zi]  32 KB
  __shared__ __bf16 sZ[64 * Z_DIM];       //                 16 KB
  __shared__ float  sBz[Z_DIM];
  const int tid = threadIdx.x, lane = tid & 31, wave = tid >> 5;
  const int hi = lane >> 4, nl = lane & 15;
  const int r0 = blockIdx.x * 64;

  for (int i = tid; i < Z_DIM * Z_DIM; i += 256) {
    int zo = i >> 7, zi = i & 127;
    sWzt[i] = (__bf16)Wz[zi * Z_DIM + zo];
  }
  for (int i = tid; i < 64 * Z_DIM; i += 256) {
    int r = i >> 7, kk = i & 127;
    sZ[i] = (__bf16)z[(size_t)(r0 + r) * Z_DIM + kk];
  }
  if (tid < Z_DIM) sBz[tid] = bz[tid];
  __syncthreads();

#pragma unroll
  for (int tt = 0; tt < 4; ++tt) {
    int t = wave * 4 + tt;
    int m0 = (t >> 3) * 16, n0 = (t & 7) * 16;
    v8f c = {};
#pragma unroll
    for (int kk = 0; kk < Z_DIM; kk += 32)
      c = wmma_bf16(c, frag_a(sZ, Z_DIM, m0, kk, lane),
                       frag_bt(sWzt, Z_DIM, n0, kk, lane));
#pragma unroll
    for (int r = 0; r < 8; ++r) {
      int row = r0 + m0 + r + hi * 8, col = n0 + nl;
      fz[(size_t)row * Z_DIM + col] = c[r] + sBz[col];
    }
  }
}

// ---------------------------------------------------------------------------
// Category bucketing
// ---------------------------------------------------------------------------
__global__ void k_init(int* counts, int* cursor) {
  int i = blockIdx.x * blockDim.x + threadIdx.x;
  if (i < C_DIM) { counts[i] = 0; cursor[i] = 0; }
}
__global__ void k_count(const int* __restrict__ c, int* counts) {
  int i = blockIdx.x * blockDim.x + threadIdx.x;
  if (i < N_TOT) atomicAdd(&counts[c[i]], 1);
}
__global__ void k_scan(const int* __restrict__ counts, int* offsets) {
  if (threadIdx.x == 0 && blockIdx.x == 0) {
    int acc = 0;
    for (int k = 0; k < C_DIM; ++k) { offsets[k] = acc; acc += counts[k]; }
  }
}
__global__ void k_scatter(const int* __restrict__ c, const int* __restrict__ offsets,
                          int* cursor, int* perm) {
  int i = blockIdx.x * blockDim.x + threadIdx.x;
  if (i < N_TOT) {
    int k = c[i];
    int pos = offsets[k] + atomicAdd(&cursor[k], 1);
    perm[pos] = i;
  }
}

// ---------------------------------------------------------------------------
// Per-category: U = Fx_k @ Ws[k]; T; masked softplus row-means; final output.
// One block per category (n_k ~ 128), 256 threads.
// ---------------------------------------------------------------------------
__global__ __launch_bounds__(256) void k_cat(const float* __restrict__ fx,
                                             const float* __restrict__ fz,
                                             const float* __restrict__ Ws,
                                             const int* __restrict__ counts,
                                             const int* __restrict__ offsets,
                                             const int* __restrict__ perm,
                                             float* __restrict__ out) {
  __shared__ __bf16 sWkt[Z_DIM * Z_DIM];  // Ws[k]^T: [e][d]  32 KB
  __shared__ __bf16 sA[64 * Z_DIM];       // gathered fx      16 KB
  __shared__ __bf16 sU[64 * Z_DIM];       // U tile (bf16)    16 KB
  __shared__ __bf16 sFz[64 * Z_DIM];      // gathered fz      16 KB
  __shared__ float  sS[64 * 65];          // softplus tile  16.25 KB
  __shared__ float  sRow[64];
  __shared__ float  sT[64];
  __shared__ int    sIdx[64];

  const int tid = threadIdx.x, lane = tid & 31, wave = tid >> 5;
  const int hi = lane >> 4, nl = lane & 15;
  const int k = blockIdx.x;
  const int n = counts[k];
  const int base = offsets[k];
  if (n == 0) return;
  const float* Wk = Ws + (size_t)k * Z_DIM * Z_DIM;

  for (int i = tid; i < Z_DIM * Z_DIM; i += 256) {
    int e = i >> 7, d = i & 127;
    sWkt[i] = (__bf16)Wk[d * Z_DIM + e];   // Bt[e][d] = Wk[d][e]
  }

  for (int i0 = 0; i0 < n; i0 += 64) {
    __syncthreads();
    if (tid < 64) {
      sIdx[tid] = (i0 + tid < n) ? perm[base + i0 + tid] : -1;
      sRow[tid] = 0.f;
    }
    __syncthreads();
    for (int i = tid; i < 64 * Z_DIM; i += 256) {
      int r = i >> 7, d = i & 127;
      int gi = sIdx[r];
      sA[i] = (gi >= 0) ? (__bf16)fx[(size_t)gi * Z_DIM + d] : (__bf16)0.f;
    }
    __syncthreads();

    // U[64][128] = sA @ Wk ; 32 tiles, 4/wave
#pragma unroll
    for (int tt = 0; tt < 4; ++tt) {
      int t = wave * 4 + tt;
      int m0 = (t >> 3) * 16, n0 = (t & 7) * 16;
      v8f c = {};
#pragma unroll
      for (int kk = 0; kk < Z_DIM; kk += 32)
        c = wmma_bf16(c, frag_a(sA, Z_DIM, m0, kk, lane),
                         frag_bt(sWkt, Z_DIM, n0, kk, lane));
#pragma unroll
      for (int r = 0; r < 8; ++r)
        sU[(m0 + r + hi * 8) * Z_DIM + (n0 + nl)] = (__bf16)c[r];
    }
    __syncthreads();

    // T_i = softplus(U_i . fz_i)
    if (tid < 64) {
      int gi = sIdx[tid];
      if (gi >= 0) {
        float acc = 0.f;
        for (int d = 0; d < Z_DIM; ++d)
          acc += (float)sU[tid * Z_DIM + d] * fz[(size_t)gi * Z_DIM + d];
        sT[tid] = softplus_f(acc);
      }
    }

    for (int j0 = 0; j0 < n; j0 += 64) {
      __syncthreads();
      for (int i = tid; i < 64 * Z_DIM; i += 256) {
        int r = i >> 7, d = i & 127;
        int gj = (j0 + r < n) ? perm[base + j0 + r] : -1;
        sFz[i] = (gj >= 0) ? (__bf16)fz[(size_t)gj * Z_DIM + d] : (__bf16)0.f;
      }
      __syncthreads();
      // S[64][64] = U @ Fz^T ; 16 tiles, 2/wave; softplus + j-mask on store
#pragma unroll
      for (int tt = 0; tt < 2; ++tt) {
        int t = wave * 2 + tt;
        int m0 = (t >> 2) * 16, n0 = (t & 3) * 16;
        v8f c = {};
#pragma unroll
        for (int kk = 0; kk < Z_DIM; kk += 32)
          c = wmma_bf16(c, frag_a(sU, Z_DIM, m0, kk, lane),
                           frag_bt(sFz, Z_DIM, n0, kk, lane));
        int jc = n0 + nl;
        bool jvalid = (j0 + jc) < n;
#pragma unroll
        for (int r = 0; r < 8; ++r)
          sS[(m0 + r + hi * 8) * 65 + jc] = jvalid ? softplus_f(c[r]) : 0.f;
      }
      __syncthreads();
      if (tid < 64) {
        float s = 0.f;
        for (int j = 0; j < 64; ++j) s += sS[tid * 65 + j];
        sRow[tid] += s;
      }
    }
    __syncthreads();
    if (tid < 64) {
      int gi = sIdx[tid];
      if (gi >= 0) {
        float neg = sRow[tid] / (float)n;
        out[gi] = logf(sT[tid] + EPSV) - logf(neg + EPSV);
      }
    }
  }
}

// ---------------------------------------------------------------------------
extern "C" void kernel_launch(void* const* d_in, const int* in_sizes, int n_in,
                              void* d_out, int out_size, void* d_ws, size_t ws_size,
                              hipStream_t stream) {
  const float* x  = (const float*)d_in[0];
  const int*   c  = (const int*)  d_in[1];
  const float* z  = (const float*)d_in[2];
  const float* W1 = (const float*)d_in[3];
  const float* b1 = (const float*)d_in[4];
  const float* W2 = (const float*)d_in[5];
  const float* b2 = (const float*)d_in[6];
  const float* Wz = (const float*)d_in[7];
  const float* bz = (const float*)d_in[8];
  const float* Ws = (const float*)d_in[9];
  float* out = (float*)d_out;

  char* ws = (char*)d_ws;
  float* fx = (float*)ws;                                           // 4 MB
  float* fz = (float*)(ws + (size_t)N_TOT * Z_DIM * sizeof(float)); // 4 MB
  int* counts  = (int*)(ws + (size_t)2 * N_TOT * Z_DIM * sizeof(float));
  int* offsets = counts + C_DIM;
  int* cursor  = offsets + C_DIM;
  int* perm    = cursor + C_DIM;

  k_init   <<<1, 64, 0, stream>>>(counts, cursor);
  k_count  <<<N_TOT / 256, 256, 0, stream>>>(c, counts);
  k_scan   <<<1, 32, 0, stream>>>(counts, offsets);
  k_scatter<<<N_TOT / 256, 256, 0, stream>>>(c, offsets, cursor, perm);
  k_fx     <<<N_TOT / 64, 256, 0, stream>>>(x, W1, b1, W2, b2, fx);
  k_fz     <<<N_TOT / 64, 256, 0, stream>>>(z, Wz, bz, fz);
  k_cat    <<<C_DIM, 256, 0, stream>>>(fx, fz, Ws, counts, offsets, perm, out);
}
